// Model_31834297598472
// MI455X (gfx1250) — compile-verified, matrix operand order checked by reading, not run on hardware
//
#include <hip/hip_runtime.h>
#include <hip/hip_bf16.h>

// ---------------------------------------------------------------------------
// 2-layer LSTM LM.  T=128 B=32 H=1024 V=32000 L=2.
// bf16 WMMA for all GEMMs.  Big GEMMs (pre-gate + decoder) are fed by the
// Tensor Data Mover: tensor_load_to_lds double-buffered slabs in LDS with
// s_wait_tensorcnt + workgroup barriers; WMMA operands come from LDS
// (conflict-free via TDM pad -> 80B row pitch).  Recurrent step = 1 tiny
// WMMA kernel per timestep (launch boundary is the h(t) dependency).
// ---------------------------------------------------------------------------

typedef __attribute__((ext_vector_type(16))) __bf16 v16bf;
typedef __attribute__((ext_vector_type(8)))  __bf16 v8bf;
typedef __attribute__((ext_vector_type(8)))  float  v8f;
typedef __attribute__((ext_vector_type(4))) unsigned int u32x4;
typedef __attribute__((ext_vector_type(4))) int          i32x4;
typedef __attribute__((ext_vector_type(8))) int          i32x8;

#define T_SEQ 128
#define B_SZ  32
#define V_SZ  32000
#define H_SZ  1024
#define L_SZ  2
#define TB    (T_SEQ * B_SZ)   // 4096
#define G4H   (4 * H_SZ)       // 4096

#if __has_builtin(__builtin_amdgcn_tensor_load_to_lds)
#define HAVE_TDM 1
#else
#define HAVE_TDM 0
#endif

// ---- LDS tiling for the big GEMMs -----------------------------------------
// Macro tile 64(M) x 256(N), K-slab 32.  LDS rows padded to 80B (TDM pad:
// every 16 DWORDs insert 4 DWORDs) => bank-conflict-free b128 reads.
#define LDS_ROW_H    40                       // halves per LDS row (80B)
#define A_TILE_BYTES (64 * 80)                // 5120
#define B_TILE_BYTES (256 * 80)               // 20480
#define BUF_BYTES    (A_TILE_BYTES + B_TILE_BYTES)  // 25600
#define GEMM_LDS_BYTES (2 * BUF_BYTES)        // 51200

// Load a 16x32 bf16 A/B fragment from a row-major matrix (row stride ld
// halves).  VGPRs 0-3 <- cols [k0,k0+16), VGPRs 4-7 <- cols [k0+16,k0+32).
__device__ __forceinline__ v16bf load_frag(const __bf16* __restrict__ base,
                                           int ld, int r0, int k0, int lane) {
  const __bf16* p = base + (size_t)(r0 + (lane & 15)) * (size_t)ld
                         + (size_t)k0 + (size_t)((lane >> 4) * 8);
  v8bf lo = *(const v8bf*)p;
  v8bf hi = *(const v8bf*)(p + 16);
  return __builtin_shufflevector(lo, hi, 0,1,2,3,4,5,6,7,8,9,10,11,12,13,14,15);
}

// Same fragment shape out of an LDS tile with LDS_ROW_H pitch.
__device__ __forceinline__ v16bf frag_lds(const __bf16* s, int halfBase, int lane) {
  const __bf16* p = s + halfBase + (lane & 15) * LDS_ROW_H + (lane >> 4) * 8;
  v8bf lo = *(const v8bf*)p;
  v8bf hi = *(const v8bf*)(p + 16);
  return __builtin_shufflevector(lo, hi, 0,1,2,3,4,5,6,7,8,9,10,11,12,13,14,15);
}

__device__ __forceinline__ v8f wmma_bf16(v16bf a, v16bf b, v8f c) {
  return __builtin_amdgcn_wmma_f32_16x16x32_bf16(false, a, false, b,
                                                 (short)0, c, false, false);
}

__device__ __forceinline__ float sigf(float x) { return 1.0f / (1.0f + __expf(-x)); }

#if HAVE_TDM
// Issue one TDM 2-D tile load: rows x 32 halves from a row-major bf16 tensor
// into LDS at lds_byte_off, LDS rows padded to 80B (pad_interval=16DW,
// pad_amount=4DW).  D# fields per CDNA5 ISA ch.8.
__device__ __forceinline__ void tdm_load_tile(const __bf16* gsrc,
                                              unsigned lds_byte_off,
                                              unsigned tensor_d0,   // elements
                                              unsigned tensor_d1,   // rows
                                              unsigned long long d0_stride, // elements
                                              unsigned tile_rows) {
  unsigned long long ga = (unsigned long long)(size_t)gsrc;
  u32x4 g0;
  g0[0] = 1u;                                           // count=1, user mode
  g0[1] = lds_byte_off;                                 // lds_addr
  g0[2] = (unsigned)ga;                                 // global_addr[31:0]
  g0[3] = (unsigned)((ga >> 32) & 0x1FFFFFFu) | (2u << 30); // [56:32] | type=2
  i32x8 g1;
  unsigned dw0 = (1u << 16)                             // data_size = 2 bytes
               | (1u << 20)                             // pad_enable
               | (3u << 22)                             // pad_interval: 16 DW
               | (3u << 25);                            // pad_amount:   4 DW
  g1[0] = (int)dw0;
  g1[1] = (int)((tensor_d0 & 0xFFFFu) << 16);           // tensor_dim0[15:0]
  g1[2] = (int)(((tensor_d0 >> 16) & 0xFFFFu) |
                ((tensor_d1 & 0xFFFFu) << 16));         // dim0 hi | dim1 lo
  g1[3] = (int)(((tensor_d1 >> 16) & 0xFFFFu) |
                (32u << 16));                           // dim1 hi | tile_dim0=32
  g1[4] = (int)(tile_rows & 0xFFFFu);                   // tile_dim1 (tile_dim2=0)
  g1[5] = (int)(unsigned)(d0_stride & 0xFFFFFFFFull);   // dim0_stride[31:0]
  g1[6] = (int)(unsigned)((d0_stride >> 32) & 0xFFFFull); // dim0_stride[47:32]
  g1[7] = 0;
  i32x4 z4 = {0, 0, 0, 0};
#if defined(__clang_major__) && (__clang_major__ >= 23)
  i32x8 z8 = {0, 0, 0, 0, 0, 0, 0, 0};
  __builtin_amdgcn_tensor_load_to_lds(g0, g1, z4, z4, z8, 0);
#else
  __builtin_amdgcn_tensor_load_to_lds(g0, g1, z4, z4, 0);
#endif
}
#endif  // HAVE_TDM

// ------------------------------- prep kernels ------------------------------

__global__ void cvt_f32_bf16_kernel(const float* __restrict__ src,
                                    __bf16* __restrict__ dst, long n) {
  long i = (long)blockIdx.x * blockDim.x + threadIdx.x;
  long stride = (long)gridDim.x * blockDim.x;
  for (; i < n; i += stride) dst[i] = (__bf16)src[i];
}

__global__ void bias_sum_kernel(const float* __restrict__ b_ih,
                                const float* __restrict__ b_hh,
                                float* __restrict__ bias, int n) {
  int i = blockIdx.x * blockDim.x + threadIdx.x;
  if (i < n) bias[i] = b_ih[i] + b_hh[i];
}

__global__ void embed_kernel(const int* __restrict__ tokens,
                             const float* __restrict__ emb,
                             __bf16* __restrict__ x, long n /* TB*H */) {
  long i = (long)blockIdx.x * blockDim.x + threadIdx.x;
  long stride = (long)gridDim.x * blockDim.x;
  for (; i < n; i += stride) {
    int row = (int)(i >> 10);          // i / H
    int h   = (int)(i & (H_SZ - 1));   // i % H
    int tok = tokens[row];
    x[i] = (__bf16)emb[(size_t)tok * H_SZ + h];
  }
}

__global__ void init_state_kernel(const float* __restrict__ h0,
                                  const float* __restrict__ c0,
                                  __bf16* __restrict__ h_bf,
                                  float* __restrict__ h_f,
                                  float* __restrict__ c_f, int n) {
  int i = blockIdx.x * blockDim.x + threadIdx.x;
  if (i < n) {
    float hv = h0[i], cv = c0[i];
    h_bf[i] = (__bf16)hv;
    h_f[i]  = hv;
    c_f[i]  = cv;
  }
}

__global__ void copy_state_kernel(const float* __restrict__ hf,
                                  const float* __restrict__ cf,
                                  float* __restrict__ outh,
                                  float* __restrict__ outc, int n) {
  int i = blockIdx.x * blockDim.x + threadIdx.x;
  if (i < n) { outh[i] = hf[i]; outc[i] = cf[i]; }
}

// ---------------------- WMMA GEMM, TDM/LDS-staged --------------------------
// out[M,N] = A[M,K] (bf16 row-major) * W[N,K]^T (bf16 row-major) + bias[N]
// Workgroup = 8 waves = 2x4 wave grid over a 64x256 macro tile.
// Per K-slab (32): TDM DMAs A(64x32) and W(256x32) into padded LDS buffers
// (double-buffered); each wave does 8 wmma from LDS.
// Requires M%64==0, N%256==0, K%32==0.
__global__ void __launch_bounds__(256) gemm_bias_lds_kernel(
    const __bf16* __restrict__ A, const __bf16* __restrict__ W,
    const float* __restrict__ bias, float* __restrict__ out,
    int M, int N, int K) {
  extern __shared__ __bf16 smem[];
  int lane = threadIdx.x & 31;
  int wave = threadIdx.x >> 5;             // 0..7
  int nMacro = N >> 8;
  int mTile0 = (int)(blockIdx.x / nMacro) * 64;
  int nTile0 = (int)(blockIdx.x % nMacro) * 256;
  int waveM = (wave >> 2) * 32;            // 0 / 32
  int waveN = (wave & 3) * 64;             // 0 / 64 / 128 / 192

  v8f acc0[4], acc1[4];
#pragma unroll
  for (int j = 0; j < 4; ++j) {
    acc0[j] = (v8f){0.f,0.f,0.f,0.f,0.f,0.f,0.f,0.f};
    acc1[j] = (v8f){0.f,0.f,0.f,0.f,0.f,0.f,0.f,0.f};
  }

  auto fetch = [&](int buf, int k) {
#if HAVE_TDM
    if (wave == 0) {
      tdm_load_tile(A + (size_t)mTile0 * K + k, (unsigned)(buf * BUF_BYTES),
                    (unsigned)K, (unsigned)M, (unsigned long long)K, 64u);
      tdm_load_tile(W + (size_t)nTile0 * K + k,
                    (unsigned)(buf * BUF_BYTES + A_TILE_BYTES),
                    (unsigned)K, (unsigned)N, (unsigned long long)K, 256u);
    }
#else
    // Cooperative fallback with identical padded LDS layout.
    for (int idx = threadIdx.x; idx < 64 * 4; idx += 256) {
      int r = idx >> 2, c = idx & 3;
      *(v8bf*)&smem[(buf * BUF_BYTES) / 2 + r * LDS_ROW_H + c * 8] =
          *(const v8bf*)(A + (size_t)(mTile0 + r) * K + k + c * 8);
    }
    for (int idx = threadIdx.x; idx < 256 * 4; idx += 256) {
      int r = idx >> 2, c = idx & 3;
      *(v8bf*)&smem[(buf * BUF_BYTES + A_TILE_BYTES) / 2 + r * LDS_ROW_H + c * 8] =
          *(const v8bf*)(W + (size_t)(nTile0 + r) * K + k + c * 8);
    }
#endif
  };

  int nk = K / 32;
  fetch(0, 0);                                   // prologue
  for (int i = 0; i < nk; ++i) {
    int cur = i & 1;
    if (i + 1 < nk) fetch(cur ^ 1, (i + 1) * 32);  // overlap next slab
#if HAVE_TDM
    if (wave == 0) {
      if (i + 1 < nk) {
        __builtin_amdgcn_s_wait_tensorcnt(2);      // slab `cur` done; next in flight
      } else {
        __builtin_amdgcn_s_wait_tensorcnt(0);      // last slab: drain
      }
    }
#endif
    __syncthreads();                               // slab `cur` ready

    int aBase = (cur * BUF_BYTES) / 2;
    int bBase = (cur * BUF_BYTES + A_TILE_BYTES) / 2;
    v16bf a0 = frag_lds(smem, aBase + (waveM     ) * LDS_ROW_H, lane);
    v16bf a1 = frag_lds(smem, aBase + (waveM + 16) * LDS_ROW_H, lane);
#pragma unroll
    for (int j = 0; j < 4; ++j) {
      v16bf b = frag_lds(smem, bBase + (waveN + j * 16) * LDS_ROW_H, lane);
      acc0[j] = wmma_bf16(a0, b, acc0[j]);
      acc1[j] = wmma_bf16(a1, b, acc1[j]);
    }
    __syncthreads();                               // done reading before refill
  }

  int cn = lane & 15;
  int mh = (lane >> 4) * 8;
#pragma unroll
  for (int j = 0; j < 4; ++j) {
    int col = nTile0 + waveN + j * 16 + cn;
    float bv = bias[col];
#pragma unroll
    for (int r = 0; r < 8; ++r) {
      out[(size_t)(mTile0 + waveM      + mh + r) * N + col] = acc0[j][r] + bv;
      out[(size_t)(mTile0 + waveM + 16 + mh + r) * N + col] = acc1[j][r] + bv;
    }
  }
}

// --------------------------- LSTM recurrent step ---------------------------
// 128 waves; wave gw: mb = gw&1 (batch rows), nb = gw>>1 (hidden block).
// acc[g] init from pre-gates, accumulate h@Whh^T via WMMA, fused cell update.
__global__ void __launch_bounds__(128) lstm_step_kernel(
    const __bf16* __restrict__ Whh,   // [4H, H]
    const float*  __restrict__ pre,   // [TB, 4H]
    const __bf16* __restrict__ h_in,  // [B, H]
    __bf16*       __restrict__ h_out, // [B, H]
    float*        __restrict__ h_f,   // [B, H]
    float*        __restrict__ c_f,   // [B, H]
    __bf16*       __restrict__ yseq,  // [TB, H]
    int t) {
  int lane = threadIdx.x & 31;
  int gw = blockIdx.x * (blockDim.x >> 5) + (threadIdx.x >> 5);  // 0..127
  int mb = (gw & 1) * 16;
  int nb = (gw >> 1) * 16;

  int cn = lane & 15;
  int mh = (lane >> 4) * 8;

  v8f acc[4];
#pragma unroll
  for (int g = 0; g < 4; ++g) {
#pragma unroll
    for (int r = 0; r < 8; ++r) {
      acc[g][r] = pre[(size_t)(t * B_SZ + mb + mh + r) * G4H + g * H_SZ + nb + cn];
    }
  }

  for (int k = 0; k < H_SZ; k += 32) {
    if (k + 32 < H_SZ)
      __builtin_prefetch(Whh + (size_t)(nb + (lane & 15)) * H_SZ + k + 32, 0, 1);
    v16bf a = load_frag(h_in, H_SZ, mb, k, lane);
#pragma unroll
    for (int g = 0; g < 4; ++g) {
      v16bf b = load_frag(Whh, H_SZ, g * H_SZ + nb, k, lane);
      acc[g] = wmma_bf16(a, b, acc[g]);
    }
  }

#pragma unroll
  for (int r = 0; r < 8; ++r) {
    int m = mb + mh + r;
    int n = nb + cn;
    size_t idx = (size_t)m * H_SZ + n;
    float ig = sigf(acc[0][r]);
    float fg = sigf(acc[1][r]);
    float gg = tanhf(acc[2][r]);
    float og = sigf(acc[3][r]);
    float c  = fg * c_f[idx] + ig * gg;
    float h  = og * tanhf(c);
    c_f[idx] = c;
    h_f[idx] = h;
    __bf16 hb = (__bf16)h;
    h_out[idx] = hb;
    yseq[(size_t)(t * B_SZ + m) * H_SZ + n] = hb;
  }
}

// ------------------------------ log-softmax --------------------------------
__global__ void __launch_bounds__(256) log_softmax_kernel(float* __restrict__ p,
                                                          int V) {
  __shared__ float red[8];
  float* row = p + (size_t)blockIdx.x * V;
  int tid = threadIdx.x, lane = tid & 31, wv = tid >> 5;

  float mx = -3.4e38f;
  for (int j = tid; j < V; j += 256) mx = fmaxf(mx, row[j]);
  for (int o = 16; o; o >>= 1) mx = fmaxf(mx, __shfl_xor(mx, o, 32));
  if (lane == 0) red[wv] = mx;
  __syncthreads();
  mx = red[0];
#pragma unroll
  for (int w = 1; w < 8; ++w) mx = fmaxf(mx, red[w]);
  __syncthreads();

  float s = 0.f;
  for (int j = tid; j < V; j += 256) s += __expf(row[j] - mx);
  for (int o = 16; o; o >>= 1) s += __shfl_xor(s, o, 32);
  if (lane == 0) red[wv] = s;
  __syncthreads();
  s = 0.f;
#pragma unroll
  for (int w = 0; w < 8; ++w) s += red[w];
  float lse = mx + __logf(s);

  for (int j = tid; j < V; j += 256) row[j] = row[j] - lse;
}

// ------------------------------- launcher ----------------------------------

extern "C" void kernel_launch(void* const* d_in, const int* in_sizes, int n_in,
                              void* d_out, int out_size, void* d_ws, size_t ws_size,
                              hipStream_t stream) {
  const int*   tokens = (const int*)  d_in[0];
  const float* h0     = (const float*)d_in[1];
  const float* c0     = (const float*)d_in[2];
  const float* emb    = (const float*)d_in[3];
  const float* w_ih   = (const float*)d_in[4];
  const float* w_hh   = (const float*)d_in[5];
  const float* b_ih   = (const float*)d_in[6];
  const float* b_hh   = (const float*)d_in[7];
  const float* w_dec  = (const float*)d_in[8];
  const float* b_dec  = (const float*)d_in[9];
  float* out = (float*)d_out;

  char* base = (char*)d_ws;
  size_t off = 0;
  auto carve = [&](size_t bytes) -> char* {
    char* p = base + off;
    off = (off + bytes + 255) & ~(size_t)255;
    return p;
  };
  __bf16* X0   = (__bf16*)carve((size_t)TB * H_SZ * 2);
  __bf16* X1   = (__bf16*)carve((size_t)TB * H_SZ * 2);
  __bf16* WIH  = (__bf16*)carve((size_t)L_SZ * G4H * H_SZ * 2);
  __bf16* WHH  = (__bf16*)carve((size_t)L_SZ * G4H * H_SZ * 2);
  __bf16* WDEC = (__bf16*)carve((size_t)V_SZ * H_SZ * 2);
  float*  BIAS = (float*) carve((size_t)L_SZ * G4H * 4);
  float*  PRE  = (float*) carve((size_t)TB * G4H * 4);
  __bf16* HBF0 = (__bf16*)carve((size_t)B_SZ * H_SZ * 2);
  __bf16* HBF1 = (__bf16*)carve((size_t)B_SZ * H_SZ * 2);
  float*  HF   = (float*) carve((size_t)B_SZ * H_SZ * 4);
  float*  CF   = (float*) carve((size_t)B_SZ * H_SZ * 4);
  (void)ws_size; (void)in_sizes; (void)n_in; (void)out_size;

  cvt_f32_bf16_kernel<<<4096, 256, 0, stream>>>(w_ih, WIH, (long)L_SZ * G4H * H_SZ);
  cvt_f32_bf16_kernel<<<4096, 256, 0, stream>>>(w_hh, WHH, (long)L_SZ * G4H * H_SZ);
  cvt_f32_bf16_kernel<<<8192, 256, 0, stream>>>(w_dec, WDEC, (long)V_SZ * H_SZ);
  bias_sum_kernel<<<(L_SZ * G4H + 255) / 256, 256, 0, stream>>>(b_ih, b_hh, BIAS, L_SZ * G4H);
  embed_kernel<<<4096, 256, 0, stream>>>(tokens, emb, X0, (long)TB * H_SZ);

  __bf16* xin = X0;
  __bf16* xout = X1;
  const int stateN = B_SZ * H_SZ;

  for (int l = 0; l < L_SZ; ++l) {
    {
      int blocks = (TB / 64) * (G4H / 256);        // 64 * 16 = 1024
      gemm_bias_lds_kernel<<<blocks, 256, GEMM_LDS_BYTES, stream>>>(
          xin, WIH + (size_t)l * G4H * H_SZ, BIAS + (size_t)l * G4H,
          PRE, TB, G4H, H_SZ);
    }
    init_state_kernel<<<(stateN + 255) / 256, 256, 0, stream>>>(
        h0 + (size_t)l * stateN, c0 + (size_t)l * stateN, HBF0, HF, CF, stateN);

    for (int t = 0; t < T_SEQ; ++t) {
      const __bf16* hin  = (t & 1) ? HBF1 : HBF0;
      __bf16*       hout = (t & 1) ? HBF0 : HBF1;
      lstm_step_kernel<<<32, 128, 0, stream>>>(
          WHH + (size_t)l * G4H * H_SZ, PRE, hin, hout, HF, CF, xout, t);
    }
    copy_state_kernel<<<(stateN + 255) / 256, 256, 0, stream>>>(
        HF, CF,
        out + (size_t)TB * V_SZ + (size_t)l * stateN,
        out + (size_t)TB * V_SZ + (size_t)L_SZ * stateN + (size_t)l * stateN,
        stateN);

    __bf16* tmp = xin; xin = xout; xout = tmp;
  }

  {
    int blocks = (TB / 64) * (V_SZ / 256);         // 64 * 125 = 8000
    gemm_bias_lds_kernel<<<blocks, 256, GEMM_LDS_BYTES, stream>>>(
        xin, WDEC, b_dec, out, TB, V_SZ, H_SZ);
  }

  log_softmax_kernel<<<TB, 256, 0, stream>>>(out, V_SZ);
}
